// DecoderLSTM_87969520156822
// MI455X (gfx1250) — compile-verified, matrix operand order checked by reading, not run on hardware
//
#include <hip/hip_runtime.h>
#include <hip/hip_bf16.h>
#include <math.h>

// Problem constants (match reference)
#define Bq   64
#define Tq   128
#define Sq   256
#define Hq   2048          // H == I == K == C
#define Gq   (4 * Hq)      // 4H
#define FLOAT_MIN_V (-100000.0f)

// ---------------------------------------------------------------------------
// WMMA types (gfx1250, wave32)
// ---------------------------------------------------------------------------
typedef __attribute__((ext_vector_type(16))) __bf16 v16bf;
typedef __attribute__((ext_vector_type(8)))  float  v8f;

struct __align__(16) U4 { unsigned int x, y, z, w; };
struct __align__(8)  U2 { unsigned int x, y; };

__device__ __forceinline__ unsigned short f2bf(float f) {
    unsigned u = __float_as_uint(f);
    unsigned r = u + 0x7fffu + ((u >> 16) & 1u);   // round-to-nearest-even
    return (unsigned short)(r >> 16);
}

__device__ __forceinline__ float bf2f(unsigned short h) {
    return __uint_as_float(((unsigned)h) << 16);
}

__device__ __forceinline__ float sigf(float x) {
    return 1.0f / (1.0f + __expf(-x));
}

// Load a 16x32 bf16 operand tile from global (A: MxK rows, or B: NxK rows of
// a row-major [rows x ld] matrix, K contiguous).  Per the CDNA5 ISA 16-bit
// operand layout: lane L<16 holds row L, K = {0..7, 16..23}; lane L>=16 holds
// row L-16, K = {8..15, 24..31}.  Two 16-byte chunks per lane.
__device__ __forceinline__ v16bf load_tile_bf16(const unsigned short* base,
                                                size_t ld, int k0, int lane) {
    int half = (lane >> 4) & 1;
    int r    = lane & 15;
    const unsigned short* p = base + (size_t)r * ld + (size_t)k0;
    union { v16bf v; U4 q[2]; } u;
    u.q[0] = *(const U4*)(p + half * 8);        // K = half*8 .. half*8+7
    u.q[1] = *(const U4*)(p + 16 + half * 8);   // K = 16+half*8 .. 16+half*8+7
    return u.v;
}

// Same operand layout, but sourced from an LDS-staged tile (tile stored as
// 16 rows x 64B, i.e. 64 16-byte chunks). L points at a 12-tile buffer.
__device__ __forceinline__ v16bf lds_tile_bf16(const U4* L, int tile,
                                               int r, int half) {
    union { v16bf v; U4 q[2]; } u;
    const U4* p = L + tile * 64 + r * 4;
    u.q[0] = p[half];       // K chunk at byte half*16
    u.q[1] = p[2 + half];   // K chunk at byte 32 + half*16
    return u.v;
}

#define WMMA_BF16(a, b, c) \
    __builtin_amdgcn_wmma_f32_16x16x32_bf16(false, (a), false, (b), (short)0, (c), false, false)

// ---------------------------------------------------------------------------
// Prep kernels
// ---------------------------------------------------------------------------
__global__ void k_f32_to_bf16_v4(const float4* __restrict__ src,
                                 U2* __restrict__ dst, int n4) {
    int i = blockIdx.x * blockDim.x + threadIdx.x;
    if (i < n4) {
        float4 v = src[i];
        U2 o;
        o.x = (unsigned)f2bf(v.x) | ((unsigned)f2bf(v.y) << 16);
        o.y = (unsigned)f2bf(v.z) | ((unsigned)f2bf(v.w) << 16);
        dst[i] = o;
    }
}

__global__ void k_zero_state(float* __restrict__ c_state,
                             unsigned short* __restrict__ h_bf, int n) {
    int i = blockIdx.x * blockDim.x + threadIdx.x;
    if (i < n) { c_state[i] = 0.0f; h_bf[i] = 0; }
}

// ---------------------------------------------------------------------------
// q = h @ W_attr^T    (out fp32 [B, Hq])
// M-blocked: each wave owns a 64x16 output strip (4 accumulators), loads the
// weight (B) tile once per k-chunk and reuses it across the 4 A tiles.
// grid: (Hq/16/8), block 256 (8 waves)
// ---------------------------------------------------------------------------
__global__ void k_q_gemm(const unsigned short* __restrict__ h_bf,
                         const unsigned short* __restrict__ Wattr_bf,
                         float* __restrict__ q_out) {
    const int lane = threadIdx.x & 31;
    const int wave = threadIdx.x >> 5;
    const int n0 = (blockIdx.x * 8 + wave) * 16;

    const unsigned short* Bbase = Wattr_bf + (size_t)n0 * Hq;

    v8f acc[4] = {{}, {}, {}, {}};
    for (int k = 0; k < Hq; k += 32) {
        if (k + 256 < Hq)
            __builtin_prefetch(Bbase + (size_t)(lane & 15) * Hq + k + 256, 0, 1);
        v16bf bt = load_tile_bf16(Bbase, Hq, k, lane);
        #pragma unroll
        for (int mi = 0; mi < 4; ++mi) {
            v16bf a = load_tile_bf16(h_bf + (size_t)(mi * 16) * Hq, Hq, k, lane);
            acc[mi] = WMMA_BF16(a, bt, acc[mi]);
        }
    }

    const int half = lane >> 4;
    const int col  = lane & 15;
    #pragma unroll
    for (int mi = 0; mi < 4; ++mi)
        #pragma unroll
        for (int i = 0; i < 8; ++i) {
            int m = mi * 16 + i + 8 * half;
            q_out[(size_t)m * Hq + n0 + col] = acc[mi][i];
        }
}

// ---------------------------------------------------------------------------
// Attention energies: e[b,s] = mask*sum_k tanh(ek+q+b_attr)*V + (1-mask)*MIN
// enc_keys pre-converted to bf16 (halves the dominant HBM stream).
// grid: B*S blocks, 256 threads; each thread owns 8 contiguous k (one b128).
// ---------------------------------------------------------------------------
__global__ void k_attn_energy(const unsigned short* __restrict__ ek_bf,
                              const float* __restrict__ q,
                              const float* __restrict__ b_attr,
                              const float* __restrict__ V_attr,
                              const float* __restrict__ enc_mask,
                              float* __restrict__ e_out) {
    __shared__ float red[256];
    const int bs = blockIdx.x;            // b*S + s
    const int b  = bs >> 8;               // S == 256
    const int k0 = threadIdx.x * 8;       // 256 threads * 8 == K == 2048

    union { U4 u; unsigned short s[8]; } kk;
    kk.u = *(const U4*)(ek_bf + (size_t)bs * Hq + k0);
    const float* qb = q      + (size_t)b * Hq + k0;
    const float* ba = b_attr + k0;
    const float* va = V_attr + k0;

    float p = 0.0f;
    #pragma unroll
    for (int i = 0; i < 8; ++i)
        p += tanhf(bf2f(kk.s[i]) + qb[i] + ba[i]) * va[i];

    red[threadIdx.x] = p;
    __syncthreads();
    for (int off = 128; off > 0; off >>= 1) {
        if ((int)threadIdx.x < off) red[threadIdx.x] += red[threadIdx.x + off];
        __syncthreads();
    }
    if (threadIdx.x == 0) {
        float m = enc_mask[bs];
        e_out[bs] = m * red[0] + (1.0f - m) * FLOAT_MIN_V;
    }
}

// ---------------------------------------------------------------------------
// Softmax over S (=256 = blockDim) per batch row
// ---------------------------------------------------------------------------
__global__ void k_softmax(const float* __restrict__ e,
                          float* __restrict__ scores) {
    __shared__ float red[256];
    const int b = blockIdx.x;
    const int s = threadIdx.x;
    const float v = e[b * Sq + s];

    red[s] = v;
    __syncthreads();
    for (int off = 128; off > 0; off >>= 1) {
        if (s < off) red[s] = fmaxf(red[s], red[s + off]);
        __syncthreads();
    }
    const float mx = red[0];
    __syncthreads();

    const float ex = __expf(v - mx);
    red[s] = ex;
    __syncthreads();
    for (int off = 128; off > 0; off >>= 1) {
        if (s < off) red[s] += red[s + off];
        __syncthreads();
    }
    scores[b * Sq + s] = ex / red[0];
}

// ---------------------------------------------------------------------------
// ctx[b,c] = sum_s scores[b,s] * enc_values[b,s,c]   (enc_values in bf16)
// Each thread owns 4 contiguous channels (8B loads per lane per s).
// Writes fp32 output slice and bf16 GEMM operand.
// grid: (C/1024, B), block 256
// ---------------------------------------------------------------------------
__global__ void k_ctx(const unsigned short* __restrict__ ev_bf,
                      const float* __restrict__ scores,
                      unsigned short* __restrict__ ctx_bf,
                      float* __restrict__ out_ctxs, int t) {
    const int b  = blockIdx.y;
    const int c0 = (blockIdx.x * 256 + threadIdx.x) * 4;
    const unsigned short* ev = ev_bf + ((size_t)b * Sq) * Hq + c0;
    const float* sc = scores + b * Sq;

    float a0 = 0.f, a1 = 0.f, a2 = 0.f, a3 = 0.f;
    #pragma unroll 4
    for (int s = 0; s < Sq; ++s) {
        union { U2 u; unsigned short h[4]; } v;
        v.u = *(const U2*)(ev + (size_t)s * Hq);
        float w = sc[s];
        a0 += w * bf2f(v.h[0]);
        a1 += w * bf2f(v.h[1]);
        a2 += w * bf2f(v.h[2]);
        a3 += w * bf2f(v.h[3]);
    }

    float* oc = out_ctxs + (((size_t)b * Tq) + t) * Hq + c0;
    oc[0] = a0; oc[1] = a1; oc[2] = a2; oc[3] = a3;

    U2 o;
    o.x = (unsigned)f2bf(a0) | ((unsigned)f2bf(a1) << 16);
    o.y = (unsigned)f2bf(a2) | ((unsigned)f2bf(a3) << 16);
    *(U2*)(ctx_bf + (size_t)b * Hq + c0) = o;
}

// ---------------------------------------------------------------------------
// gates = y_t @ W_y^T + h @ U_h^T + ctx @ C_ctx^T + b      (fp32 [B, 4H])
//
// Block = 256 threads (8 waves); each wave owns a 64x16 output strip.
// The 12 A tiles per k-chunk (4 M-tiles x {y_t, h, ctx}) are SHARED by all
// 8 waves -> staged once per block into double-buffered LDS (2 x 12KB):
//   * 256 threads cooperatively fetch chunk k+1 (3 x b128 each) while the
//     WMMAs consume chunk k from LDS (ds_load_b128 pairs, short latency),
//   * weight (B) tiles stay per-wave in registers from global (+prefetch),
//   * one s_barrier per k-chunk instead of a loadcnt-0 wait per WMMA.
// grid: (Gq/16/8), block 256
// ---------------------------------------------------------------------------
__global__ void k_gates_gemm(const unsigned short* __restrict__ y_bf, int t,
                             const unsigned short* __restrict__ h_bf,
                             const unsigned short* __restrict__ ctx_bf,
                             const unsigned short* __restrict__ Wy_bf,
                             const unsigned short* __restrict__ Uh_bf,
                             const unsigned short* __restrict__ Cc_bf,
                             const float* __restrict__ bias,
                             float* __restrict__ gates) {
    __shared__ U4 lds[2][12 * 64];   // 12 tiles x 64 x 16B = 12KB per buffer

    const int tid  = threadIdx.x;
    const int lane = tid & 31;
    const int wave = tid >> 5;
    const int n0 = (blockIdx.x * 8 + wave) * 16;

    // --- cooperative A-stage addressing: 3 x 16B chunks per thread ----------
    const unsigned short* gsrc[3];
    int ldst[3];
    #pragma unroll
    for (int j = 0; j < 3; ++j) {
        int chunk = tid + j * 256;       // 0..767
        int tile  = chunk >> 6;          // 0..11 : tile = o*4 + mi
        int w     = chunk & 63;
        int r     = w >> 2;              // row within tile (0..15)
        int c4    = w & 3;               // 16B chunk within row (0..3)
        int o     = tile >> 2;           // 0=y_t, 1=h, 2=ctx
        int mi    = tile & 3;
        int m     = mi * 16 + r;         // global M row
        const unsigned short* g;
        if (o == 0)      g = y_bf   + ((size_t)m * Tq + t) * Hq;
        else if (o == 1) g = h_bf   + (size_t)m * Hq;
        else             g = ctx_bf + (size_t)m * Hq;
        gsrc[j] = g + c4 * 8;            // add k at use time
        ldst[j] = tile * 64 + r * 4 + c4;
    }

    // --- per-wave weight (B) bases ------------------------------------------
    const unsigned short* Bw = Wy_bf + (size_t)n0 * Hq;
    const unsigned short* Bu = Uh_bf + (size_t)n0 * Hq;
    const unsigned short* Bc = Cc_bf + (size_t)n0 * Hq;

    const int r    = lane & 15;
    const int half = lane >> 4;

    // --- prologue: stage k-chunk 0 ------------------------------------------
    #pragma unroll
    for (int j = 0; j < 3; ++j)
        lds[0][ldst[j]] = *(const U4*)(gsrc[j]);
    __syncthreads();

    v8f acc[4] = {{}, {}, {}, {}};
    const int NK = Hq / 32;
    for (int kc = 0; kc < NK; ++kc) {
        const int k   = kc * 32;
        const int cur = kc & 1;
        const bool more = (kc + 1) < NK;

        // fetch next A chunk into registers (overlaps with WMMAs below)
        U4 nv[3];
        if (more) {
            #pragma unroll
            for (int j = 0; j < 3; ++j)
                nv[j] = *(const U4*)(gsrc[j] + (size_t)(k + 32));
        }

        // per-wave weight tiles from global (L2-resident stream)
        if (k + 256 < Hq) {
            __builtin_prefetch(Bw + (size_t)r * Hq + k + 256, 0, 1);
            __builtin_prefetch(Bu + (size_t)r * Hq + k + 256, 0, 1);
            __builtin_prefetch(Bc + (size_t)r * Hq + k + 256, 0, 1);
        }
        v16bf bw = load_tile_bf16(Bw, Hq, k, lane);
        v16bf bu = load_tile_bf16(Bu, Hq, k, lane);
        v16bf bc = load_tile_bf16(Bc, Hq, k, lane);

        // 12 WMMAs from LDS-staged A tiles
        const U4* L = &lds[cur][0];
        #pragma unroll
        for (int mi = 0; mi < 4; ++mi) {
            v16bf ay = lds_tile_bf16(L, 0 + mi, r, half);
            acc[mi] = WMMA_BF16(ay, bw, acc[mi]);
            v16bf ah = lds_tile_bf16(L, 4 + mi, r, half);
            acc[mi] = WMMA_BF16(ah, bu, acc[mi]);
            v16bf ac = lds_tile_bf16(L, 8 + mi, r, half);
            acc[mi] = WMMA_BF16(ac, bc, acc[mi]);
        }

        // publish next chunk to the other buffer
        if (more) {
            U4* Ln = &lds[cur ^ 1][0];
            #pragma unroll
            for (int j = 0; j < 3; ++j)
                Ln[ldst[j]] = nv[j];
        }
        __syncthreads();
    }

    // --- epilogue: bias + store ---------------------------------------------
    const float bb = bias[n0 + r];
    #pragma unroll
    for (int mi = 0; mi < 4; ++mi)
        #pragma unroll
        for (int i = 0; i < 8; ++i) {
            int m = mi * 16 + i + 8 * half;
            gates[(size_t)m * Gq + n0 + r] = acc[mi][i] + bb;
        }
}

// ---------------------------------------------------------------------------
// LSTM pointwise update; writes c state, h bf16 state, and hs output slice.
// ---------------------------------------------------------------------------
__global__ void k_lstm_update(const float* __restrict__ gates,
                              float* __restrict__ c_state,
                              unsigned short* __restrict__ h_bf,
                              float* __restrict__ out_hs, int t) {
    const int idx = blockIdx.x * 256 + threadIdx.x;   // 0 .. B*H-1
    const int b = idx >> 11;                          // Hq == 2048
    const int j = idx & (Hq - 1);
    const float* g = gates + (size_t)b * Gq;

    const float gi = g[j];
    const float gf = g[Hq + j];
    const float go = g[2 * Hq + j];
    const float gc = g[3 * Hq + j];

    const float c_new = sigf(gi) * tanhf(gc) + sigf(gf) * c_state[idx];
    const float h_new = sigf(go) * tanhf(c_new);

    c_state[idx] = c_new;
    h_bf[idx]    = f2bf(h_new);
    out_hs[(((size_t)b * Tq) + t) * Hq + j] = h_new;
}

// ---------------------------------------------------------------------------
// Host-side orchestration
// ---------------------------------------------------------------------------
static inline size_t align_up(size_t x, size_t a) { return (x + a - 1) & ~(a - 1); }

extern "C" void kernel_launch(void* const* d_in, const int* in_sizes, int n_in,
                              void* d_out, int out_size, void* d_ws, size_t ws_size,
                              hipStream_t stream) {
    (void)in_sizes; (void)n_in; (void)out_size; (void)ws_size;

    const float* enc_keys   = (const float*)d_in[0];
    const float* enc_values = (const float*)d_in[1];
    const float* enc_mask   = (const float*)d_in[2];
    const float* y          = (const float*)d_in[3];
    const float* W_attr     = (const float*)d_in[4];
    const float* V_attr     = (const float*)d_in[5];
    const float* b_attr     = (const float*)d_in[6];
    const float* W_y        = (const float*)d_in[7];
    const float* U_h        = (const float*)d_in[8];
    const float* C_ctx      = (const float*)d_in[9];
    const float* bias       = (const float*)d_in[10];

    float* out_hs   = (float*)d_out;
    float* out_ctxs = out_hs + (size_t)Bq * Tq * Hq;

    // Workspace carve (256-byte aligned slices)
    char* p = (char*)d_ws;
    size_t off = 0;
    auto carve = [&](size_t bytes) {
        char* r = p + off;
        off = align_up(off + bytes, 256);
        return r;
    };
    unsigned short* Wattr_bf = (unsigned short*)carve((size_t)Hq * Hq * 2);
    unsigned short* Wy_bf    = (unsigned short*)carve((size_t)Gq * Hq * 2);
    unsigned short* Uh_bf    = (unsigned short*)carve((size_t)Gq * Hq * 2);
    unsigned short* Cc_bf    = (unsigned short*)carve((size_t)Gq * Hq * 2);
    unsigned short* y_bf     = (unsigned short*)carve((size_t)Bq * Tq * Hq * 2);
    unsigned short* ek_bf    = (unsigned short*)carve((size_t)Bq * Sq * Hq * 2);
    unsigned short* ev_bf    = (unsigned short*)carve((size_t)Bq * Sq * Hq * 2);
    unsigned short* h_bf     = (unsigned short*)carve((size_t)Bq * Hq * 2);
    unsigned short* ctx_bf   = (unsigned short*)carve((size_t)Bq * Hq * 2);
    float* c_state = (float*)carve((size_t)Bq * Hq * 4);
    float* q_buf   = (float*)carve((size_t)Bq * Hq * 4);
    float* e_buf   = (float*)carve((size_t)Bq * Sq * 4);
    float* s_buf   = (float*)carve((size_t)Bq * Sq * 4);
    float* gates   = (float*)carve((size_t)Bq * Gq * 4);

    // --- one-time prep: weights + streams to bf16, zero state ---------------
    auto conv = [&](const float* src, unsigned short* dst, size_t n) {
        size_t n4 = n / 4;
        k_f32_to_bf16_v4<<<dim3((unsigned)((n4 + 255) / 256)), dim3(256), 0, stream>>>(
            (const float4*)src, (U2*)dst, (int)n4);
    };
    conv(W_attr,     Wattr_bf, (size_t)Hq * Hq);
    conv(W_y,        Wy_bf,    (size_t)Gq * Hq);
    conv(U_h,        Uh_bf,    (size_t)Gq * Hq);
    conv(C_ctx,      Cc_bf,    (size_t)Gq * Hq);
    conv(y,          y_bf,     (size_t)Bq * Tq * Hq);
    conv(enc_keys,   ek_bf,    (size_t)Bq * Sq * Hq);
    conv(enc_values, ev_bf,    (size_t)Bq * Sq * Hq);

    k_zero_state<<<dim3((Bq * Hq) / 256), dim3(256), 0, stream>>>(
        c_state, h_bf, Bq * Hq);

    // --- sequential scan over T --------------------------------------------
    for (int t = 0; t < Tq; ++t) {
        // q = h @ W_attr^T
        k_q_gemm<<<dim3(Hq / 16 / 8), dim3(256), 0, stream>>>(
            h_bf, Wattr_bf, q_buf);

        // attention energies + mask
        k_attn_energy<<<dim3(Bq * Sq), dim3(256), 0, stream>>>(
            ek_bf, q_buf, b_attr, V_attr, enc_mask, e_buf);

        // softmax over S
        k_softmax<<<dim3(Bq), dim3(Sq), 0, stream>>>(e_buf, s_buf);

        // ctx = scores^T . enc_values  (+ write ctxs output slice)
        k_ctx<<<dim3(Hq / 1024, Bq), dim3(256), 0, stream>>>(
            ev_bf, s_buf, ctx_bf, out_ctxs, t);

        // gates = y_t W_y^T + h U_h^T + ctx C_ctx^T + b
        k_gates_gemm<<<dim3(Gq / 16 / 8), dim3(256), 0, stream>>>(
            y_bf, t, h_bf, ctx_bf, Wy_bf, Uh_bf, Cc_bf, bias, gates);

        // pointwise LSTM update (+ write hs output slice)
        k_lstm_update<<<dim3((Bq * Hq) / 256), dim3(256), 0, stream>>>(
            gates, c_state, h_bf, out_hs, t);
    }
}
